// LocallyConnected2d_33191507263939
// MI455X (gfx1250) — matrix operand order repaired and checked by reading, compiled, not run on hardware
//
#include <hip/hip_runtime.h>

// ---------------------------------------------------------------------------
// LocallyConnected2d: B=64, CIN=64, COUT=64, H=W=32, K=3, stride=1, pad=1
// Per location (h,w): GEMM  [64 batch] x [576 = cin*9] x [64 cout]
// Strategy: 1 block per location, 512 threads = 16 wave32s, each wave one
// 16x16 f32 accumulator tile, fed by v_wmma_f32_16x16x32_bf16 (K chunked
// 3 x 192, one filter-row per chunk so global loads are 12B-contiguous).
// ---------------------------------------------------------------------------

typedef __attribute__((ext_vector_type(16))) __bf16 v16bf;
typedef __attribute__((ext_vector_type(8)))  float  v8f;
typedef unsigned short u16;
typedef __attribute__((ext_vector_type(8)))  u16    v8u;
typedef __attribute__((ext_vector_type(16))) u16    v16u;

static __device__ __forceinline__ u16 f2bf(float f) {
  // round-to-nearest-even f32 -> bf16 (bits)
  unsigned int u = __builtin_bit_cast(unsigned int, f);
  u += 0x7FFFu + ((u >> 16) & 1u);
  return (u16)(u >> 16);
}

#define KP 200  // padded K stride (halfs) per LDS row: 192 data + 8 pad

__global__ __launch_bounds__(512)
void lc2d_wmma_kernel(const float* __restrict__ x,
                      const float* __restrict__ wgt,
                      const float* __restrict__ bias,
                      float* __restrict__ out) {
  // A: patches[b][kk], B: weights[o][kk]; kk = i*3 + kw within a chunk
  __shared__ __align__(16) u16 Ash[64 * KP];
  __shared__ __align__(16) u16 Bsh[64 * KP];

  const int loc = blockIdx.x;          // 0..1023
  const int h   = loc >> 5;
  const int w   = loc & 31;

  const int tid    = threadIdx.x;
  const int lane   = tid & 31;
  const int waveId = tid >> 5;         // 0..15
  const int mt     = waveId >> 2;      // batch-tile 0..3
  const int nt     = waveId & 3;       // cout-tile  0..3
  const int g      = lane >> 4;        // lane group 0/1
  const int ln     = lane & 15;

  const int rowA = mt * 16 + ln;       // batch row this lane reads for A
  const int colB = nt * 16 + ln;       // cout col this lane reads for B

  v8f acc = {};

  for (int kh = 0; kh < 3; ++kh) {
    // ---- stage chunk kh: taps (kh, kw=0..2) for all (b,i) and (i,o) ----
    const int xrow = h + kh - 1;
    const bool row_ok = (unsigned)xrow < 32u;
    #pragma unroll
    for (int r = 0; r < 8; ++r) {
      const int p  = (r << 9) + tid;   // 0..4095
      const int hi = p >> 6;           // A: batch b   | B: cin i
      const int lo = p & 63;           // A: cin i     | B: cout o

      // A: x_pad[b=hi, i=lo, xrow, w-1..w+1]  (12B contiguous, masked edges)
      float v0 = 0.f, v1 = 0.f, v2 = 0.f;
      if (row_ok) {
        const float* xr = x + ((((hi << 6) + lo) * 32 + xrow) << 5);
        if (w > 0)  v0 = xr[w - 1];
        v1 = xr[w];
        if (w < 31) v2 = xr[w + 1];
      }
      u16* ap = &Ash[hi * KP + lo * 3];
      ap[0] = f2bf(v0); ap[1] = f2bf(v1); ap[2] = f2bf(v2);

      // B: wgt[i=hi, o=lo, h, w, kh, 0..2]  (12B contiguous)
      const float* wp = wgt + (p * 1024 + loc) * 9 + kh * 3;
      u16* bp = &Bsh[lo * KP + hi * 3];
      bp[0] = f2bf(wp[0]); bp[1] = f2bf(wp[1]); bp[2] = f2bf(wp[2]);
    }
    __syncthreads();

    // ---- 6 WMMA K-steps over this 192-wide chunk ----
    const v8u* arow = reinterpret_cast<const v8u*>(&Ash[rowA * KP]);
    const v8u* brow = reinterpret_cast<const v8u*>(&Bsh[colB * KP]);
    #pragma unroll
    for (int s = 0; s < 6; ++s) {
      // A 16x32 bf16 fragment: e<8 -> K=32s+8g+e ; e>=8 -> K=32s+16+8g+(e-8)
      v8u alo = arow[4 * s + g];
      v8u ahi = arow[4 * s + 2 + g];
      // B 32x16 bf16 fragment: e<8 -> K=32s+8g+e ; e>=8 -> K=32s+16+8g+(e-8)
      v8u blo = brow[4 * s + g];
      v8u bhi = brow[4 * s + 2 + g];
      v16u au = __builtin_shufflevector(alo, ahi,
                 0,1,2,3,4,5,6,7,8,9,10,11,12,13,14,15);
      v16u bu = __builtin_shufflevector(blo, bhi,
                 0,1,2,3,4,5,6,7,8,9,10,11,12,13,14,15);
      v16bf av = __builtin_bit_cast(v16bf, au);
      v16bf bv = __builtin_bit_cast(v16bf, bu);
      acc = __builtin_amdgcn_wmma_f32_16x16x32_bf16(
              /*neg_a=*/false, av, /*neg_b=*/false, bv,
              /*c_mod=*/(short)0, acc, /*reuse_a=*/false, /*reuse_b=*/false);
    }
    __syncthreads();
  }

  // ---- epilogue: D VGPR r -> (M = r + 8g, N = ln); add bias, store ----
  const int o_idx = nt * 16 + ln;
  const float bv = bias[o_idx * 1024 + loc];
  #pragma unroll
  for (int r = 0; r < 8; ++r) {
    const int b_idx = mt * 16 + g * 8 + r;
    out[(b_idx * 64 + o_idx) * 1024 + loc] = acc[r] + bv;
  }
}

extern "C" void kernel_launch(void* const* d_in, const int* in_sizes, int n_in,
                              void* d_out, int out_size, void* d_ws, size_t ws_size,
                              hipStream_t stream) {
  const float* x    = (const float*)d_in[0];
  const float* wgt  = (const float*)d_in[1];
  const float* bias = (const float*)d_in[2];
  float* out = (float*)d_out;
  (void)in_sizes; (void)n_in; (void)out_size; (void)d_ws; (void)ws_size;

  dim3 grid(1024);   // one block per output location (h,w)
  dim3 block(512);   // 16 wave32s -> 4x4 tiles of 16x16 covering 64x64
  hipLaunchKernelGGL(lc2d_wmma_kernel, grid, block, 0, stream, x, wgt, bias, out);
}
